// RecurrentGate_20023137534722
// MI455X (gfx1250) — matrix operand order repaired and checked by reading, not compiled, not run on hardware
//
#include <hip/hip_runtime.h>

// ---------------------------------------------------------------------------
// RecurrentGate (GRU-style gate) for MI455X / gfx1250.
//   H=1024, M=B*S=32768, K=2H=2048. Three GEMMs [32768x2048]@[2048x1024]^T.
//   Kernel 1: z = sigmoid([x,h]Wg^T+bg), r = sigmoid([x,h]Wu^T+bu)
//             -> ws: z (bf16), r*h_prev (bf16)
//   Kernel 2: hc = tanh([x, r*h]Wc^T+bc); out = (1-z)*h + z*hc
// Compute-bound (~412 GFLOP vs <1 GB traffic @23.3TB/s) -> bf16 WMMA, f32 acc.
// Weights (24MB fp32) stay L2-resident (192MB). Double-buffered LDS pipeline
// hides global latency behind the v_wmma chain. Workspace: 128 MB.
// ---------------------------------------------------------------------------

typedef __attribute__((ext_vector_type(4)))  __bf16 v4bf;
typedef __attribute__((ext_vector_type(8)))  __bf16 v8bf;
typedef __attribute__((ext_vector_type(16))) __bf16 v16bf;
typedef __attribute__((ext_vector_type(8)))  float  v8f;

#define HDIM   1024
#define KDIM   2048
#define MTOT   32768
#define BM     128
#define BN     64
#define KC     32
#define LDA    40                 // padded LDS row stride (bf16): 80B rows
#define A_ELEMS (BM * LDA)
#define B_ELEMS (BN * LDA)

__device__ __forceinline__ void cvt_store4(__bf16* dst, float4 v) {
    v4bf t = { (__bf16)v.x, (__bf16)v.y, (__bf16)v.z, (__bf16)v.w };
    *(v4bf*)dst = t;               // 8B ds_store, native f32->bf16 cvt
}
__device__ __forceinline__ float sigmoidf_fast(float x) {
    return 1.f / (1.f + __expf(-x));
}
__device__ __forceinline__ float tanhf_fast(float x) {
    float e = __expf(2.f * x);
    return 1.f - 2.f / (e + 1.f);
}

// A fragment per ISA 16-bit 16x32 layout: lanes0-15 M=lane K=0-7,16-23;
// lanes16-31 M=lane-16 K=8-15,24-31.
__device__ __forceinline__ v16bf load_A_frag(const __bf16* As, int row, int hi) {
    union { v16bf v; v8bf h[2]; } t;
    const int kb = hi << 3;
    t.h[0] = *(const v8bf*)&As[row * LDA + kb];
    t.h[1] = *(const v8bf*)&As[row * LDA + kb + 16];
    return t.v;
}
// B fragment (32x16): lanes0-15 col=lane K=0-7,16-23; lanes16-31 col=lane-16 K=8-15,24-31.
__device__ __forceinline__ v16bf load_B_frag(const __bf16* Bs, int col, int hi) {
    union { v16bf v; v8bf h[2]; } t;
    const int kb = hi << 3;
    t.h[0] = *(const v8bf*)&Bs[col * LDA + kb];
    t.h[1] = *(const v8bf*)&Bs[col * LDA + kb + 16];
    return t.v;
}

// ------------------------- Kernel 1: gates z, r -----------------------------
__global__ __launch_bounds__(256)
void rg_gate_kernel(const float* __restrict__ x, const float* __restrict__ h_prev,
                    const float* __restrict__ Wg, const float* __restrict__ bg,
                    const float* __restrict__ Wu, const float* __restrict__ bu,
                    __bf16* __restrict__ z_ws, __bf16* __restrict__ rh_ws) {
    __shared__ __bf16 As[2 * A_ELEMS];
    __shared__ __bf16 Bgs[2 * B_ELEMS];
    __shared__ __bf16 Bus[2 * B_ELEMS];

    const int tid  = threadIdx.x;
    const int lane = tid & 31;
    const int wave = tid >> 5;
    const int wm   = wave & 3;          // 4 waves along M
    const int wn   = wave >> 2;         // 2 waves along N
    const int m0   = blockIdx.y * BM;
    const int n0   = blockIdx.x * BN;
    const int lrow = lane & 15;
    const int hi   = lane >> 4;
    const int srow = tid >> 3;          // staging row base 0..31
    const int sc4  = (tid & 7) << 2;    // staging col (float4 granule)

    float4 rA[4], rBg[2], rBu[2];

    // ---- prologue: prefetch + stage chunk 0 (k0 = 0, source = x) ----
#pragma unroll
    for (int it = 0; it < 4; ++it)
        rA[it] = *(const float4*)(x + (size_t)(m0 + srow + it * 32) * HDIM + sc4);
#pragma unroll
    for (int it = 0; it < 2; ++it) {
        const size_t off = (size_t)(n0 + srow + it * 32) * KDIM + sc4;
        rBg[it] = *(const float4*)(Wg + off);
        rBu[it] = *(const float4*)(Wu + off);
    }
#pragma unroll
    for (int it = 0; it < 4; ++it)
        cvt_store4(&As[(srow + it * 32) * LDA + sc4], rA[it]);
#pragma unroll
    for (int it = 0; it < 2; ++it) {
        cvt_store4(&Bgs[(srow + it * 32) * LDA + sc4], rBg[it]);
        cvt_store4(&Bus[(srow + it * 32) * LDA + sc4], rBu[it]);
    }
    __syncthreads();

    v8f accg[2][2] = {};
    v8f accu[2][2] = {};

    for (int k0 = 0; k0 < KDIM; k0 += KC) {
        const int cur = (k0 / KC) & 1;
        const int nxt = cur ^ 1;
        const bool has_next = (k0 + KC) < KDIM;

        // ---- issue global prefetch of chunk k0+KC (hidden behind WMMAs) ----
        if (has_next) {
            const int kn = k0 + KC;
            const float* srcA = (kn < HDIM)
                ? (x + (size_t)m0 * HDIM + kn)
                : (h_prev + (size_t)m0 * HDIM + (kn - HDIM));
#pragma unroll
            for (int it = 0; it < 4; ++it)
                rA[it] = *(const float4*)(srcA + (size_t)(srow + it * 32) * HDIM + sc4);
#pragma unroll
            for (int it = 0; it < 2; ++it) {
                const size_t off = (size_t)(n0 + srow + it * 32) * KDIM + kn + sc4;
                rBg[it] = *(const float4*)(Wg + off);
                rBu[it] = *(const float4*)(Wu + off);
            }
        }

        // ---- compute on buffer `cur` ----
        const __bf16* Ac  = As  + cur * A_ELEMS;
        const __bf16* Bgc = Bgs + cur * B_ELEMS;
        const __bf16* Buc = Bus + cur * B_ELEMS;
        v16bf a[2], fg[2], fu[2];
#pragma unroll
        for (int mi = 0; mi < 2; ++mi)
            a[mi] = load_A_frag(Ac, wm * 32 + mi * 16 + lrow, hi);
#pragma unroll
        for (int ni = 0; ni < 2; ++ni) {
            const int col = wn * 32 + ni * 16 + lrow;
            fg[ni] = load_B_frag(Bgc, col, hi);
            fu[ni] = load_B_frag(Buc, col, hi);
        }
#pragma unroll
        for (int mi = 0; mi < 2; ++mi)
#pragma unroll
            for (int ni = 0; ni < 2; ++ni) {
                accg[mi][ni] = __builtin_amdgcn_wmma_f32_16x16x32_bf16(
                    false, a[mi], false, fg[ni], (short)0, accg[mi][ni], false, false);
                accu[mi][ni] = __builtin_amdgcn_wmma_f32_16x16x32_bf16(
                    false, a[mi], false, fu[ni], (short)0, accu[mi][ni], false, false);
            }

        // ---- convert + stage chunk k0+KC into buffer `nxt` ----
        if (has_next) {
            __bf16* An  = As  + nxt * A_ELEMS;
            __bf16* Bgn = Bgs + nxt * B_ELEMS;
            __bf16* Bun = Bus + nxt * B_ELEMS;
#pragma unroll
            for (int it = 0; it < 4; ++it)
                cvt_store4(&An[(srow + it * 32) * LDA + sc4], rA[it]);
#pragma unroll
            for (int it = 0; it < 2; ++it) {
                cvt_store4(&Bgn[(srow + it * 32) * LDA + sc4], rBg[it]);
                cvt_store4(&Bun[(srow + it * 32) * LDA + sc4], rBu[it]);
            }
        }
        __syncthreads();
    }

    // ---- epilogue: sigmoid gates -> z, r*h_prev (bf16 workspace) ----
#pragma unroll
    for (int ni = 0; ni < 2; ++ni) {
        const int n = n0 + wn * 32 + ni * 16 + lrow;
        const float biasg = bg[n];
        const float biasu = bu[n];
#pragma unroll
        for (int mi = 0; mi < 2; ++mi) {
            const int mbase = m0 + wm * 32 + mi * 16 + hi * 8;
#pragma unroll
            for (int r = 0; r < 8; ++r) {
                const size_t o = (size_t)(mbase + r) * HDIM + n;
                const float z  = sigmoidf_fast(accg[mi][ni][r] + biasg);
                const float rr = sigmoidf_fast(accu[mi][ni][r] + biasu);
                z_ws[o]  = (__bf16)z;
                rh_ws[o] = (__bf16)(rr * h_prev[o]);
            }
        }
    }
}

// --------------------- Kernel 2: candidate + blend --------------------------
__global__ __launch_bounds__(256)
void rg_cand_kernel(const float* __restrict__ x, const float* __restrict__ h_prev,
                    const __bf16* __restrict__ rh_ws,
                    const float* __restrict__ Wc, const float* __restrict__ bc,
                    const __bf16* __restrict__ z_ws, float* __restrict__ out) {
    __shared__ __bf16 As[2 * A_ELEMS];
    __shared__ __bf16 Bcs[2 * B_ELEMS];

    const int tid  = threadIdx.x;
    const int lane = tid & 31;
    const int wave = tid >> 5;
    const int wm   = wave & 3;
    const int wn   = wave >> 2;
    const int m0   = blockIdx.y * BM;
    const int n0   = blockIdx.x * BN;
    const int lrow = lane & 15;
    const int hi   = lane >> 4;
    const int srow = tid >> 3;
    const int sc4  = (tid & 7) << 2;

    float4 rA[4];      // fp32 prefetch (k < H: from x)
    uint2  rR[4];      // bf16 prefetch (k >= H: from r*h workspace)
    float4 rBc[2];

    // ---- prologue: chunk 0 (from x) ----
#pragma unroll
    for (int it = 0; it < 4; ++it)
        rA[it] = *(const float4*)(x + (size_t)(m0 + srow + it * 32) * HDIM + sc4);
#pragma unroll
    for (int it = 0; it < 2; ++it)
        rBc[it] = *(const float4*)(Wc + (size_t)(n0 + srow + it * 32) * KDIM + sc4);
#pragma unroll
    for (int it = 0; it < 4; ++it)
        cvt_store4(&As[(srow + it * 32) * LDA + sc4], rA[it]);
#pragma unroll
    for (int it = 0; it < 2; ++it)
        cvt_store4(&Bcs[(srow + it * 32) * LDA + sc4], rBc[it]);
    __syncthreads();

    v8f acc[2][2] = {};

    for (int k0 = 0; k0 < KDIM; k0 += KC) {
        const int cur = (k0 / KC) & 1;
        const int nxt = cur ^ 1;
        const bool has_next   = (k0 + KC) < KDIM;
        const bool next_is_x  = (k0 + KC) < HDIM;

        // ---- prefetch chunk k0+KC ----
        if (has_next) {
            const int kn = k0 + KC;
            if (next_is_x) {
#pragma unroll
                for (int it = 0; it < 4; ++it)
                    rA[it] = *(const float4*)(x + (size_t)(m0 + srow + it * 32) * HDIM
                                              + kn + sc4);
            } else {
                const __bf16* src = rh_ws + (size_t)m0 * HDIM + (kn - HDIM);
#pragma unroll
                for (int it = 0; it < 4; ++it)
                    rR[it] = *(const uint2*)(src + (size_t)(srow + it * 32) * HDIM + sc4);
            }
#pragma unroll
            for (int it = 0; it < 2; ++it)
                rBc[it] = *(const float4*)(Wc + (size_t)(n0 + srow + it * 32) * KDIM
                                           + kn + sc4);
        }

        // ---- compute on buffer `cur` ----
        const __bf16* Ac  = As  + cur * A_ELEMS;
        const __bf16* Bcc = Bcs + cur * B_ELEMS;
        v16bf a[2], fb[2];
#pragma unroll
        for (int mi = 0; mi < 2; ++mi)
            a[mi] = load_A_frag(Ac, wm * 32 + mi * 16 + lrow, hi);
#pragma unroll
        for (int ni = 0; ni < 2; ++ni)
            fb[ni] = load_B_frag(Bcc, wn * 32 + ni * 16 + lrow, hi);
#pragma unroll
        for (int mi = 0; mi < 2; ++mi)
#pragma unroll
            for (int ni = 0; ni < 2; ++ni)
                acc[mi][ni] = __builtin_amdgcn_wmma_f32_16x16x32_bf16(
                    false, a[mi], false, fb[ni], (short)0, acc[mi][ni], false, false);

        // ---- stage chunk k0+KC into buffer `nxt` ----
        if (has_next) {
            __bf16* An  = As  + nxt * A_ELEMS;
            __bf16* Bcn = Bcs + nxt * B_ELEMS;
            if (next_is_x) {
#pragma unroll
                for (int it = 0; it < 4; ++it)
                    cvt_store4(&An[(srow + it * 32) * LDA + sc4], rA[it]);
            } else {
#pragma unroll
                for (int it = 0; it < 4; ++it)
                    *(uint2*)&An[(srow + it * 32) * LDA + sc4] = rR[it];
            }
#pragma unroll
            for (int it = 0; it < 2; ++it)
                cvt_store4(&Bcn[(srow + it * 32) * LDA + sc4], rBc[it]);
        }
        __syncthreads();
    }

    // ---- epilogue: tanh candidate, blend with z, fp32 out ----
#pragma unroll
    for (int ni = 0; ni < 2; ++ni) {
        const int n = n0 + wn * 32 + ni * 16 + lrow;
        const float biasc = bc[n];
#pragma unroll
        for (int mi = 0; mi < 2; ++mi) {
            const int mbase = m0 + wm * 32 + mi * 16 + hi * 8;
#pragma unroll
            for (int r = 0; r < 8; ++r) {
                const size_t o = (size_t)(mbase + r) * HDIM + n;
                const float hc = tanhf_fast(acc[mi][ni][r] + biasc);
                const float z  = (float)z_ws[o];
                out[o] = (1.f - z) * h_prev[o] + z * hc;
            }
        }
    }
}

// ---------------------------------------------------------------------------
extern "C" void kernel_launch(void* const* d_in, const int* in_sizes, int n_in,
                              void* d_out, int out_size, void* d_ws, size_t ws_size,
                              hipStream_t stream) {
    const float* x  = (const float*)d_in[0];
    const float* hp = (const float*)d_in[1];
    const float* Wg = (const float*)d_in[2];
    const float* bg = (const float*)d_in[3];
    const float* Wu = (const float*)d_in[4];
    const float* bu = (const float*)d_in[5];
    const float* Wc = (const float*)d_in[6];
    const float* bc = (const float*)d_in[7];
    float* out = (float*)d_out;

    __bf16* z_ws  = (__bf16*)d_ws;                       // 64 MB
    __bf16* rh_ws = z_ws + (size_t)MTOT * HDIM;          // 64 MB

    dim3 grid(HDIM / BN, MTOT / BM);   // 16 x 256 blocks, 256 thr (8 waves)
    rg_gate_kernel<<<grid, 256, 0, stream>>>(x, hp, Wg, bg, Wu, bu, z_ws, rh_ws);
    rg_cand_kernel<<<grid, 256, 0, stream>>>(x, hp, rh_ws, Wc, bc, z_ws, out);
}
